// LSTMSimple_46566035423844
// MI455X (gfx1250) — matrix operand order repaired and checked by reading, not compiled, not run on hardware
//
#include <hip/hip_runtime.h>
#include <hip/hip_bf16.h>

typedef __bf16 bf16_t;
typedef __attribute__((ext_vector_type(8)))  bf16_t v8bf;
typedef __attribute__((ext_vector_type(16))) bf16_t v16bf;
typedef __attribute__((ext_vector_type(8)))  float  v8f;
typedef __attribute__((ext_vector_type(4)))  float  v4f;

#define T_STEPS 8192
#define I_DIM   1024
#define H_DIM   1024
#define G_DIM   4096          // 4*H
#define NB      256           // persistent recurrence blocks (each owns 4 hidden units)

// ---- workspace layout (bytes) ----
#define GATES_OFF   ((size_t)0)
#define GATES_BYTES ((size_t)T_STEPS * G_DIM * 4)      // 128 MiB fp32 gates_x
#define XB_OFF      (GATES_OFF + GATES_BYTES)
#define XB_BYTES    ((size_t)T_STEPS * I_DIM * 2)      // 16 MiB bf16 x
#define WB_OFF      (XB_OFF + XB_BYTES)
#define WB_BYTES    ((size_t)G_DIM * I_DIM * 2)        // 8 MiB bf16 W_ih
#define BIAS_OFF    (WB_OFF + WB_BYTES)
#define BIAS_BYTES  ((size_t)G_DIM * 4)
#define H_OFF       (BIAS_OFF + BIAS_BYTES)
#define H_BYTES     ((size_t)H_DIM * 4)
#define CNT_OFF     (H_OFF + H_BYTES)

// ---------------- fp32 -> bf16 conversion ----------------
__global__ __launch_bounds__(256) void cvt_bf16(const float* __restrict__ src,
                                                bf16_t* __restrict__ dst, int n) {
    int i = blockIdx.x * 256 + threadIdx.x;
    if (i < n) dst[i] = (bf16_t)src[i];
}

// ---------------- bias sum, h init, counter zero ----------------
__global__ __launch_bounds__(256) void prep(const float* __restrict__ b_ih,
                                            const float* __restrict__ b_hh,
                                            const float* __restrict__ h0,
                                            float* __restrict__ bias,
                                            float* __restrict__ hbuf,
                                            unsigned* __restrict__ cnt) {
    int i = blockIdx.x * 256 + threadIdx.x;
    if (i < G_DIM) bias[i] = b_ih[i] + b_hh[i];
    if (i < H_DIM) hbuf[i] = h0[i];
    if (i == 0)    *cnt = 0u;
}

// ---------------- phase 1: gates_x = x @ W_ih^T + bias (WMMA bf16) ----------------
// grid (T/16, 8), block 256 = 8 waves. Each wave computes a 16(M) x 64(N) tile:
// A fragment loaded once per K-step and reused across 4 WMMAs (2.5 b128 loads/WMMA).
__global__ __launch_bounds__(256) void gemm_gates(const bf16_t* __restrict__ xb,
                                                  const bf16_t* __restrict__ wb,
                                                  const float*  __restrict__ bias,
                                                  float*        __restrict__ gates) {
    const int wave = threadIdx.x >> 5;
    const int lane = threadIdx.x & 31;
    const int half = lane >> 4;         // which 16-lane group
    const int r    = lane & 15;
    const int m0 = blockIdx.x * 16;
    const int n0 = (blockIdx.y * 8 + wave) * 64;   // 64-wide N slab per wave

    // A: row (m0+r) of x; B tile nt: column (n0+nt*16+r) of W_ih^T == that row of W_ih
    const bf16_t* arow = xb + (size_t)(m0 + r) * I_DIM;
    const bf16_t* brow0 = wb + (size_t)(n0 +  0 + r) * I_DIM;
    const bf16_t* brow1 = wb + (size_t)(n0 + 16 + r) * I_DIM;
    const bf16_t* brow2 = wb + (size_t)(n0 + 32 + r) * I_DIM;
    const bf16_t* brow3 = wb + (size_t)(n0 + 48 + r) * I_DIM;

    v8f acc0 = {}, acc1 = {}, acc2 = {}, acc3 = {};
#pragma unroll 2
    for (int k0 = 0; k0 < I_DIM; k0 += 32) {
        // A 16x32 bf16 per-lane layout: elems 0..15 = K k0+half*16+0..15 of row (lane&15)
        v8bf a_lo = *(const v8bf*)(arow + k0 + half * 16);
        v8bf a_hi = *(const v8bf*)(arow + k0 + half * 16 + 8);
        v16bf a = __builtin_shufflevector(a_lo, a_hi, 0,1,2,3,4,5,6,7,8,9,10,11,12,13,14,15);

        // B 32x16 bf16 per-lane layout: elems 0..15 = K k0+half*16+0..15 of column (lane&15)
        {
            v8bf b_lo = *(const v8bf*)(brow0 + k0 + half * 16);
            v8bf b_hi = *(const v8bf*)(brow0 + k0 + half * 16 + 8);
            v16bf b = __builtin_shufflevector(b_lo, b_hi, 0,1,2,3,4,5,6,7,8,9,10,11,12,13,14,15);
            acc0 = __builtin_amdgcn_wmma_f32_16x16x32_bf16(false, a, false, b, (short)0, acc0, false, false);
        }
        {
            v8bf b_lo = *(const v8bf*)(brow1 + k0 + half * 16);
            v8bf b_hi = *(const v8bf*)(brow1 + k0 + half * 16 + 8);
            v16bf b = __builtin_shufflevector(b_lo, b_hi, 0,1,2,3,4,5,6,7,8,9,10,11,12,13,14,15);
            acc1 = __builtin_amdgcn_wmma_f32_16x16x32_bf16(false, a, false, b, (short)0, acc1, false, false);
        }
        {
            v8bf b_lo = *(const v8bf*)(brow2 + k0 + half * 16);
            v8bf b_hi = *(const v8bf*)(brow2 + k0 + half * 16 + 8);
            v16bf b = __builtin_shufflevector(b_lo, b_hi, 0,1,2,3,4,5,6,7,8,9,10,11,12,13,14,15);
            acc2 = __builtin_amdgcn_wmma_f32_16x16x32_bf16(false, a, false, b, (short)0, acc2, false, false);
        }
        {
            v8bf b_lo = *(const v8bf*)(brow3 + k0 + half * 16);
            v8bf b_hi = *(const v8bf*)(brow3 + k0 + half * 16 + 8);
            v16bf b = __builtin_shufflevector(b_lo, b_hi, 0,1,2,3,4,5,6,7,8,9,10,11,12,13,14,15);
            acc3 = __builtin_amdgcn_wmma_f32_16x16x32_bf16(false, a, false, b, (short)0, acc3, false, false);
        }
    }

    // D layout: elem e = (M = m0 + e + half*8, N = ntile_base + (lane&15))
    const int rowbase = m0 + half * 8;
#pragma unroll
    for (int nt = 0; nt < 4; ++nt) {
        const v8f acc = (nt == 0) ? acc0 : (nt == 1) ? acc1 : (nt == 2) ? acc2 : acc3;
        const int col = n0 + nt * 16 + r;
        const float bv = bias[col];
#pragma unroll
        for (int e = 0; e < 8; ++e)
            gates[(size_t)(rowbase + e) * G_DIM + col] = acc[e] + bv;
    }
}

// ---------------- phase 2: persistent sequential LSTM recurrence ----------------
// NB=256 blocks x 256 threads. Block b owns hidden units n = b*4..b*4+3 (all 4 gates:
// 16 rows of W_hh, kept hot in L2). Per step: broadcast h via LDS, 2 fp32 dot rows
// per wave with shuffle reduction, threads 0..3 apply gate math, grid-wide barrier.
__global__ __launch_bounds__(256) void lstm_recur(const float* __restrict__ whh,
                                                  const float* __restrict__ gates,
                                                  const float* __restrict__ c0,
                                                  float*       h_glob,
                                                  float*       __restrict__ out,
                                                  unsigned*    counter) {
    __shared__ __align__(16) float h_lds[H_DIM];
    __shared__ float g_s[16];

    const int tid  = threadIdx.x;
    const int wave = tid >> 5;
    const int lane = tid & 31;
    const int b    = blockIdx.x;

    float c_reg = 0.0f;
    if (tid < 4) c_reg = c0[b * 4 + tid];

    // local row index li in [0,16): gate = li>>2, q = li&3 ; global row = gate*H + b*4 + q
    const int li0 = wave * 2;
    const float* wrow0 = whh + (size_t)((li0 >> 2) * H_DIM + b * 4 + (li0 & 3)) * H_DIM;
    const float* wrow1 = whh + (size_t)(((li0 + 1) >> 2) * H_DIM + b * 4 + ((li0 + 1) & 3)) * H_DIM;

    const volatile float* h_vol = (const volatile float*)h_glob;

    for (int t = 0; t < T_STEPS; ++t) {
        // broadcast current h into LDS
        for (int i = tid; i < H_DIM; i += 256) h_lds[i] = h_vol[i];
        __syncthreads();

#pragma unroll
        for (int rr = 0; rr < 2; ++rr) {
            const float* wr = rr ? wrow1 : wrow0;
            float acc = 0.0f;
#pragma unroll
            for (int i = 0; i < 8; ++i) {
                const int k = lane * 4 + i * 128;        // coalesced float4 per lane
                v4f hv = *(const v4f*)(h_lds + k);
                v4f wv = *(const v4f*)(wr + k);
                acc = fmaf(hv[0], wv[0], acc);
                acc = fmaf(hv[1], wv[1], acc);
                acc = fmaf(hv[2], wv[2], acc);
                acc = fmaf(hv[3], wv[3], acc);
            }
#pragma unroll
            for (int off = 16; off > 0; off >>= 1)
                acc += __shfl_xor(acc, off, 32);
            if (lane == 0) g_s[li0 + rr] = acc;
        }
        __syncthreads();

        if (tid < 4) {
            const int    n  = b * 4 + tid;
            const size_t gb = (size_t)t * G_DIM + n;
            const float gi = g_s[tid]      + gates[gb];
            const float gf = g_s[4 + tid]  + gates[gb + 1 * H_DIM];
            const float gg = g_s[8 + tid]  + gates[gb + 2 * H_DIM];
            const float go = g_s[12 + tid] + gates[gb + 3 * H_DIM];
            const float iv = 1.0f / (1.0f + __expf(-gi));
            const float fv = 1.0f / (1.0f + __expf(-gf));
            const float gv = tanhf(gg);
            const float ov = 1.0f / (1.0f + __expf(-go));
            c_reg = fv * c_reg + iv * gv;
            const float hn = ov * tanhf(c_reg);
            h_glob[n] = hn;
            out[(size_t)t * H_DIM + n] = hn;
            __threadfence();
        }
        __syncthreads();

        if (t + 1 < T_STEPS) {                       // device-wide split barrier
            if (tid == 0) {
                atomicAdd(counter, 1u);
                const unsigned target = (unsigned)(t + 1) * NB;
                while (__hip_atomic_load(counter, __ATOMIC_ACQUIRE,
                                         __HIP_MEMORY_SCOPE_AGENT) < target) {
                    __builtin_amdgcn_s_sleep(2);
                }
            }
            __syncthreads();
        }
    }
}

extern "C" void kernel_launch(void* const* d_in, const int* in_sizes, int n_in,
                              void* d_out, int out_size, void* d_ws, size_t ws_size,
                              hipStream_t stream) {
    const float* x    = (const float*)d_in[0];
    const float* W_ih = (const float*)d_in[1];
    const float* W_hh = (const float*)d_in[2];
    const float* b_ih = (const float*)d_in[3];
    const float* b_hh = (const float*)d_in[4];
    const float* h0   = (const float*)d_in[5];
    const float* c0   = (const float*)d_in[6];
    float* out = (float*)d_out;

    char* ws = (char*)d_ws;
    float*    gates = (float*)   (ws + GATES_OFF);
    bf16_t*   xb    = (bf16_t*)  (ws + XB_OFF);
    bf16_t*   wb    = (bf16_t*)  (ws + WB_OFF);
    float*    bias  = (float*)   (ws + BIAS_OFF);
    float*    hbuf  = (float*)   (ws + H_OFF);
    unsigned* cnt   = (unsigned*)(ws + CNT_OFF);

    cvt_bf16<<<(T_STEPS * I_DIM) / 256, 256, 0, stream>>>(x,    xb, T_STEPS * I_DIM);
    cvt_bf16<<<(G_DIM   * I_DIM) / 256, 256, 0, stream>>>(W_ih, wb, G_DIM * I_DIM);
    prep<<<16, 256, 0, stream>>>(b_ih, b_hh, h0, bias, hbuf, cnt);

    gemm_gates<<<dim3(T_STEPS / 16, G_DIM / 512), 256, 0, stream>>>(xb, wb, bias, gates);

    lstm_recur<<<NB, 256, 0, stream>>>(W_hh, gates, c0, hbuf, out, cnt);
}
